// model_8864812499567
// MI455X (gfx1250) — matrix operand order repaired and checked by reading, not compiled, not run on hardware
//
#include <hip/hip_runtime.h>
#include <hip/hip_bf16.h>
#include <cmath>
#include <cstdint>

// ---------------------------------------------------------------------------
// Sizes from the reference
// ---------------------------------------------------------------------------
#define BB   32      // batch
#define TT   100     // time steps
#define HH   512     // hidden
#define RR   60      // keys
#define EE   40      // entities
#define NHD  4       // heads
#define DKD  128     // head dim
#define TGTN 20000   // target vocab
#define OUTW (TGTN + EE)   // 20040 output row width
#define MROWS (BB * TT)    // 3200 rows of l
#define MPAD  3328         // 13 * 256
#define NPAD  20224        // 79 * 256

typedef __attribute__((ext_vector_type(16))) _Float16 v16h;
typedef __attribute__((ext_vector_type(8)))  _Float16 v8h;
typedef __attribute__((ext_vector_type(8)))  float    v8f;

__device__ __forceinline__ float sigm(float x) { return 1.0f / (1.0f + __expf(-x)); }

// Build a v16h WMMA fragment from two contiguous 16B chunks of LDS.
__device__ __forceinline__ v16h make_frag(const _Float16* lo, const _Float16* hi) {
    v8h a = *(const v8h*)lo;
    v8h b = *(const v8h*)hi;
    v16h f;
#pragma unroll
    for (int j = 0; j < 8; ++j) { f[j] = a[j]; f[8 + j] = b[j]; }
    return f;
}

// CDNA5 async global->LDS copy (16B per lane, ASYNCcnt-tracked).
__device__ __forceinline__ void async_b128(unsigned lds_off, const void* gptr) {
    asm volatile("global_load_async_to_lds_b128 %0, %1, off"
                 :: "v"(lds_off), "v"((unsigned long long)(uintptr_t)gptr)
                 : "memory");
}
__device__ __forceinline__ void wait_async0() {
    asm volatile("s_wait_asynccnt 0x0" ::: "memory");
}

// One BK=32 K-step: 2x2 WMMA tiles (32x32 per wave).
__device__ __forceinline__ void wmma_step(const _Float16 (*As)[32],
                                          const _Float16 (*Bs)[32],
                                          int wm, int wn, int lane, v8f c[2][2]) {
    const int kb = (lane >> 4) * 8;
    const int kh = (lane >> 4) * 16;
    const int lr = lane & 15;
    v16h fa[2], fb[2];
#pragma unroll
    for (int mt = 0; mt < 2; ++mt) {
        const _Float16* row = As[wm * 32 + mt * 16 + lr];
        fa[mt] = make_frag(row + kb, row + 16 + kb);
    }
#pragma unroll
    for (int nt = 0; nt < 2; ++nt) {
        const _Float16* row = Bs[wn * 32 + nt * 16 + lr];
        fb[nt] = make_frag(row + kb, row + 16 + kb);
    }
#pragma unroll
    for (int mt = 0; mt < 2; ++mt)
#pragma unroll
        for (int nt = 0; nt < 2; ++nt)
            c[mt][nt] = __builtin_amdgcn_wmma_f32_16x16x32_f16(
                false, fa[mt], false, fb[nt], (short)0, c[mt][nt], false, false);
}

// ---------------------------------------------------------------------------
// Generic f32-in/f32-out GEMM with f16 WMMA inner product (small/medium GEMMs).
// ---------------------------------------------------------------------------
template <int WROWS>
__global__ __launch_bounds__(WROWS * 128)
void gemm_f16_kernel(const float* __restrict__ A, int lda,
                     const float* __restrict__ B, int ldb,
                     const float* __restrict__ bias,
                     float* __restrict__ C, long long ldc,
                     float* __restrict__ Caux, long long ldaux,
                     int M, int N, int K) {
    constexpr int BM = WROWS * 32;
    constexpr int BN = 128;
    constexpr int NT = WROWS * 128;
    __shared__ alignas(16) _Float16 As[BM][32];
    __shared__ alignas(16) _Float16 Bs[BN][32];

    const int tid  = threadIdx.x;
    const int lane = tid & 31;
    const int wave = tid >> 5;
    const int wm   = wave >> 2;
    const int wn   = wave & 3;
    const int m0   = blockIdx.y * BM;
    const int n0   = blockIdx.x * BN;

    v8f c[2][2];
#pragma unroll
    for (int mt = 0; mt < 2; ++mt)
#pragma unroll
        for (int nt = 0; nt < 2; ++nt)
#pragma unroll
            for (int j = 0; j < 8; ++j) c[mt][nt][j] = 0.0f;

    for (int k0 = 0; k0 < K; k0 += 32) {
        for (int idx = tid; idx < BM * 32; idx += NT) {
            int m = idx >> 5, kk = idx & 31;
            int gm = m0 + m;
            float v = (gm < M) ? A[(long long)gm * lda + k0 + kk] : 0.0f;
            As[m][kk] = (_Float16)v;
        }
        for (int idx = tid; idx < BN * 32; idx += NT) {
            int kk = idx >> 7, n = idx & 127;
            int gn = n0 + n;
            float v = (gn < N) ? B[(long long)(k0 + kk) * ldb + gn] : 0.0f;
            Bs[n][kk] = (_Float16)v;
        }
        __syncthreads();
        wmma_step(As, Bs, wm, wn, lane, c);
        __syncthreads();
    }

#pragma unroll
    for (int mt = 0; mt < 2; ++mt)
#pragma unroll
        for (int nt = 0; nt < 2; ++nt)
#pragma unroll
            for (int v = 0; v < 8; ++v) {
                int gm = m0 + wm * 32 + mt * 16 + v + 8 * (lane >> 4);
                int gn = n0 + wn * 32 + nt * 16 + (lane & 15);
                if (gm < M && gn < N) {
                    float val = c[mt][nt][v] + (bias ? bias[gn] : 0.0f);
                    C[(long long)gm * ldc + gn] = val;
                    if (Caux) Caux[(long long)gm * ldaux + gn] = val;
                }
            }
}

// ---------------------------------------------------------------------------
// Heavyweight GEMM: C[M,N] = Ah[M,K](f16) @ Bt[N,K]^T(f16) + bias.
// Block 256x256 (16 waves, 64x64/wave => 16 WMMA per k-step per wave).
// Double-buffered LDS tiles filled by async global->LDS DMA (ASYNCcnt).
// Ah rows and Bt rows are contiguous in K (Bt is pre-transposed W_out).
// ---------------------------------------------------------------------------
__global__ __launch_bounds__(512)
void gemm_async_f16_kernel(const _Float16* __restrict__ Ah,   // [>=gridY*256][K]
                           const _Float16* __restrict__ Bt,   // [>=gridX*256][K]
                           const float* __restrict__ bias,
                           float* __restrict__ C, long long ldc,
                           int M, int N, int K) {
    __shared__ alignas(16) _Float16 As[2][256][32];
    __shared__ alignas(16) _Float16 Bs[2][256][32];
    const int tid  = threadIdx.x;
    const int lane = tid & 31;
    const int wave = tid >> 5;        // 0..15
    const int wm   = wave >> 2;
    const int wn   = wave & 3;
    const int m0   = blockIdx.y * 256;
    const int n0   = blockIdx.x * 256;

    v8f c[4][4];
#pragma unroll
    for (int mt = 0; mt < 4; ++mt)
#pragma unroll
        for (int nt = 0; nt < 4; ++nt)
#pragma unroll
            for (int j = 0; j < 8; ++j) c[mt][nt][j] = 0.0f;

    // Issue the 2x16KB tile DMA for k-slice k0 into buffer `buf`.
    auto issue = [&](int buf, int k0) {
#pragma unroll
        for (int j = 0; j < 2; ++j) {
            int i = tid + j * 512;            // 1024 x 16B chunks per matrix
            int row = i >> 2, seg = i & 3;    // 4 chunks per 64B LDS row
            async_b128((unsigned)(uintptr_t)&As[buf][row][seg * 8],
                       Ah + (size_t)(m0 + row) * K + k0 + seg * 8);
            async_b128((unsigned)(uintptr_t)&Bs[buf][row][seg * 8],
                       Bt + (size_t)(n0 + row) * K + k0 + seg * 8);
        }
    };

    issue(0, 0);
    wait_async0();
    __syncthreads();

    const int KS = K >> 5;
    for (int ks = 0; ks < KS; ++ks) {
        const int cur = ks & 1;
        if (ks + 1 < KS) issue(cur ^ 1, (ks + 1) * 32);   // DMA next tile now

        const _Float16 (*Ac)[32] = As[cur];
        const _Float16 (*Bc)[32] = Bs[cur];
        const int kb = (lane >> 4) * 8;
        const int kh = (lane >> 4) * 16;
        const int lr = lane & 15;
        v16h fa[4], fb[4];
#pragma unroll
        for (int mt = 0; mt < 4; ++mt) {
            const _Float16* row = Ac[wm * 64 + mt * 16 + lr];
            fa[mt] = make_frag(row + kb, row + 16 + kb);
        }
#pragma unroll
        for (int nt = 0; nt < 4; ++nt) {
            const _Float16* row = Bc[wn * 64 + nt * 16 + lr];
            fb[nt] = make_frag(row + kb, row + 16 + kb);
        }
#pragma unroll
        for (int mt = 0; mt < 4; ++mt)
#pragma unroll
            for (int nt = 0; nt < 4; ++nt)
                c[mt][nt] = __builtin_amdgcn_wmma_f32_16x16x32_f16(
                    false, fa[mt], false, fb[nt], (short)0, c[mt][nt], false, false);

        wait_async0();
        __syncthreads();
    }

#pragma unroll
    for (int mt = 0; mt < 4; ++mt)
#pragma unroll
        for (int nt = 0; nt < 4; ++nt)
#pragma unroll
            for (int v = 0; v < 8; ++v) {
                int gm = m0 + wm * 64 + mt * 16 + v + 8 * (lane >> 4);
                int gn = n0 + wn * 64 + nt * 16 + (lane & 15);
                if (gm < M && gn < N)
                    C[(long long)gm * ldc + gn] = c[mt][nt][v] + bias[gn];
            }
}

// ---------------------------------------------------------------------------
// Prep: W_out[K=1024][20000] f32 -> Wt[NPAD][1024] f16 (transposed, 0-padded)
// ---------------------------------------------------------------------------
__global__ __launch_bounds__(256)
void transpose_f16_kernel(const float* __restrict__ W, _Float16* __restrict__ Wt) {
    __shared__ float tile[32][33];
    const int kb = blockIdx.y * 32;   // K blocks: 1024/32
    const int nb = blockIdx.x * 32;   // N blocks: NPAD/32
    const int tx = threadIdx.x & 31, ty = threadIdx.x >> 5;
    for (int r = ty; r < 32; r += 8) {
        int n = nb + tx;
        tile[r][tx] = (n < TGTN) ? W[(long long)(kb + r) * TGTN + n] : 0.0f;
    }
    __syncthreads();
    for (int r = ty; r < 32; r += 8)
        Wt[(long long)(nb + r) * 1024 + kb + tx] = (_Float16)tile[tx][r];
}

// l (f32, MROWS x 1024) -> lh (f16, MPAD x 1024), zero pad rows
__global__ __launch_bounds__(256)
void cvt_f16_kernel(const float* __restrict__ src, _Float16* __restrict__ dst) {
    long long i = (long long)blockIdx.x * 256 + threadIdx.x;
    if (i < (long long)MPAD * 1024)
        dst[i] = (i < (long long)MROWS * 1024) ? (_Float16)src[i] : (_Float16)0.0f;
}

// ---------------------------------------------------------------------------
// Gate GEMM (per step): gates[32,2048] = [a | e_t | hx] @ [W_ih^T ; W_hh^T] + b
// ---------------------------------------------------------------------------
__global__ __launch_bounds__(128)
void gates_kernel(const float* __restrict__ a_prev,   // [32,512]
                  const float* __restrict__ hx_prev,  // [32,512]
                  const float* __restrict__ emb,      // [NTOKS,512]
                  const int*   __restrict__ tokens,   // [32,100]
                  int t,
                  const float* __restrict__ W_ih,     // [2048,1024]
                  const float* __restrict__ W_hh,     // [2048,512]
                  const float* __restrict__ b_ih,
                  const float* __restrict__ b_hh,
                  float* __restrict__ gates) {        // [32,2048]
    __shared__ alignas(16) _Float16 As[32][32];
    __shared__ alignas(16) _Float16 Bs[128][32];
    const int tid = threadIdx.x, lane = tid & 31, wn = tid >> 5;
    const int n0 = blockIdx.x * 128;

    v8f c[2][2];
#pragma unroll
    for (int mt = 0; mt < 2; ++mt)
#pragma unroll
        for (int nt = 0; nt < 2; ++nt)
#pragma unroll
            for (int j = 0; j < 8; ++j) c[mt][nt][j] = 0.0f;

    for (int ks = 0; ks < 48; ++ks) {
        const int k0  = ks * 32;
        const int seg = k0 >> 9;        // 0: a, 1: emb, 2: hx
        const int kl0 = k0 & 511;
        for (int idx = tid; idx < 32 * 32; idx += 128) {
            int m = idx >> 5, kk = idx & 31;
            float v;
            if (seg == 0)      v = a_prev[m * 512 + kl0 + kk];
            else if (seg == 1) v = emb[(long long)tokens[m * TT + t] * 512 + kl0 + kk];
            else               v = hx_prev[m * 512 + kl0 + kk];
            As[m][kk] = (_Float16)v;
        }
        for (int idx = tid; idx < 128 * 32; idx += 128) {
            int n = idx >> 5, kk = idx & 31;   // contiguous along K (weight rows)
            int gn = n0 + n;
            float v = (seg < 2) ? W_ih[(long long)gn * 1024 + k0 + kk]
                                : W_hh[(long long)gn * 512 + (k0 - 1024) + kk];
            Bs[n][kk] = (_Float16)v;
        }
        __syncthreads();
        wmma_step(As, Bs, 0, wn, lane, c);
        __syncthreads();
    }

#pragma unroll
    for (int mt = 0; mt < 2; ++mt)
#pragma unroll
        for (int nt = 0; nt < 2; ++nt)
#pragma unroll
            for (int v = 0; v < 8; ++v) {
                int gm = mt * 16 + v + 8 * (lane >> 4);
                int gn = n0 + wn * 32 + nt * 16 + (lane & 15);
                gates[gm * 2048 + gn] = c[mt][nt][v] + b_ih[gn] + b_hh[gn];
            }
}

// ---------------------------------------------------------------------------
// Fused LSTM elementwise + q = hx @ Wq + bq.
// ---------------------------------------------------------------------------
__global__ __launch_bounds__(128)
void lstmq_kernel(const float* __restrict__ gates,    // [32,2048]
                  const float* __restrict__ cx_prev,  // [32,512]
                  float* __restrict__ hx_new,
                  float* __restrict__ cx_new,
                  float* __restrict__ l_row,          // l + t*1024
                  const float* __restrict__ Wq,       // [512,512]
                  const float* __restrict__ bq,
                  float* __restrict__ q) {            // [32,512]
    __shared__ alignas(16) _Float16 As[32][32];
    __shared__ alignas(16) _Float16 Bs[128][32];
    const int tid = threadIdx.x, lane = tid & 31, wn = tid >> 5;
    const int n0 = blockIdx.x * 128;
    const bool writer = (blockIdx.x == 0);

    v8f c[2][2];
#pragma unroll
    for (int mt = 0; mt < 2; ++mt)
#pragma unroll
        for (int nt = 0; nt < 2; ++nt)
#pragma unroll
            for (int j = 0; j < 8; ++j) c[mt][nt][j] = 0.0f;

    for (int ks = 0; ks < 16; ++ks) {
        const int k0 = ks * 32;
        for (int idx = tid; idx < 32 * 32; idx += 128) {
            int m = idx >> 5, kk = idx & 31, j = k0 + kk;
            float iv = gates[m * 2048 + j];
            float fv = gates[m * 2048 + 512 + j];
            float gv = gates[m * 2048 + 1024 + j];
            float ov = gates[m * 2048 + 1536 + j];
            float cn = sigm(fv) * cx_prev[m * 512 + j] + sigm(iv) * tanhf(gv);
            float h  = sigm(ov) * tanhf(cn);
            As[m][kk] = (_Float16)h;
            if (writer) {
                cx_new[m * 512 + j] = cn;
                hx_new[m * 512 + j] = h;
                l_row[(long long)m * (TT * 2 * HH) + j] = h;
            }
        }
        for (int idx = tid; idx < 128 * 32; idx += 128) {
            int kk = idx >> 7, n = idx & 127;
            Bs[n][kk] = (_Float16)Wq[(long long)(k0 + kk) * 512 + n0 + n];
        }
        __syncthreads();
        wmma_step(As, Bs, 0, wn, lane, c);
        __syncthreads();
    }

#pragma unroll
    for (int mt = 0; mt < 2; ++mt)
#pragma unroll
        for (int nt = 0; nt < 2; ++nt)
#pragma unroll
            for (int v = 0; v < 8; ++v) {
                int gm = mt * 16 + v + 8 * (lane >> 4);
                int gn = n0 + wn * 32 + nt * 16 + (lane & 15);
                q[gm * 512 + gn] = c[mt][nt][v] + bq[gn];
            }
}

// ---------------------------------------------------------------------------
// 1-query masked MHA: block per (b, head).
// ---------------------------------------------------------------------------
__global__ __launch_bounds__(128)
void attn_kernel(const float* __restrict__ q, const float* __restrict__ Kp,
                 const float* __restrict__ Vp, const int* __restrict__ klens,
                 float* __restrict__ ctx) {
    __shared__ float sc[64];
    __shared__ float w[64];
    const int b = blockIdx.x >> 2;
    const int h = blockIdx.x & 3;
    const int tid = threadIdx.x;
    const int klen = klens[b];
    const float scale = 0.08838834764831845f;   // 1/sqrt(128)
    const float* qrow = q + b * 512 + h * 128;

    if (tid < RR) {
        const float* krow = Kp + (long long)(b * RR + tid) * 512 + h * 128;
        float s = 0.0f;
        for (int d = 0; d < DKD; ++d) s += qrow[d] * krow[d];
        sc[tid] = (tid < klen) ? s * scale : -1e9f;
    }
    __syncthreads();
    if (tid == 0) {
        float mx = -1e30f;
        for (int r = 0; r < RR; ++r) mx = fmaxf(mx, sc[r]);
        float sum = 0.0f;
        for (int r = 0; r < RR; ++r) { w[r] = __expf(sc[r] - mx); sum += w[r]; }
        float inv = 1.0f / sum;
        for (int r = 0; r < RR; ++r) w[r] *= inv;
    }
    __syncthreads();
    float acc = 0.0f;
    for (int r = 0; r < RR; ++r)
        acc += w[r] * Vp[(long long)(b * RR + r) * 512 + h * 128 + tid];
    ctx[b * 512 + h * 128 + tid] = acc;
}

// ---------------------------------------------------------------------------
// pgen[b*T+t] = sigmoid(l . W_switch + b_switch)
// ---------------------------------------------------------------------------
__global__ __launch_bounds__(256)
void pgen_kernel(const float* __restrict__ l, const float* __restrict__ Wsw,
                 const float* __restrict__ bsw, float* __restrict__ pgen) {
    __shared__ float red[256];
    const int row = blockIdx.x, tid = threadIdx.x;
    const float* lr = l + (long long)row * (2 * HH);
    float s = 0.0f;
    for (int k = tid; k < 2 * HH; k += 256) s += lr[k] * Wsw[k];
    red[tid] = s; __syncthreads();
    for (int st = 128; st > 0; st >>= 1) {
        if (tid < st) red[tid] += red[tid + st];
        __syncthreads();
    }
    if (tid == 0) pgen[row] = sigm(red[0] + bsw[0]);
}

// ---------------------------------------------------------------------------
// In-place row softmax over the 20000 logits already in d_out, then
// out = log(softmax * pgen + 1e-6)
// ---------------------------------------------------------------------------
__global__ __launch_bounds__(256)
void gen_softmax_kernel(float* __restrict__ out, const float* __restrict__ pgen) {
    __shared__ float red[256];
    const int row = blockIdx.x, tid = threadIdx.x;
    float* p = out + (long long)row * OUTW;

    float mx = -1e30f;
    for (int i = tid; i < TGTN; i += 256) mx = fmaxf(mx, p[i]);
    red[tid] = mx; __syncthreads();
    for (int st = 128; st > 0; st >>= 1) {
        if (tid < st) red[tid] = fmaxf(red[tid], red[tid + st]);
        __syncthreads();
    }
    const float rowmax = red[0]; __syncthreads();

    float s = 0.0f;
    for (int i = tid; i < TGTN; i += 256) s += __expf(p[i] - rowmax);
    red[tid] = s; __syncthreads();
    for (int st = 128; st > 0; st >>= 1) {
        if (tid < st) red[tid] += red[tid + st];
        __syncthreads();
    }
    const float inv = pgen[row] / red[0];
    for (int i = tid; i < TGTN; i += 256)
        p[i] = logf(__expf(p[i] - rowmax) * inv + 1e-6f);
}

// ---------------------------------------------------------------------------
// Copy-attention tail
// ---------------------------------------------------------------------------
__global__ __launch_bounds__(256)
void copy_attn_kernel(const float* __restrict__ proj, const float* __restrict__ ents,
                      const int* __restrict__ entlens, const float* __restrict__ pgen,
                      float* __restrict__ out) {
    __shared__ float pr[HH];
    __shared__ float sc[EE];
    __shared__ float mx_s, sum_s;
    const int row = blockIdx.x;           // b*T + t
    const int b = row / TT;
    const int tid = threadIdx.x;

    for (int i = tid; i < HH; i += 256) pr[i] = proj[(long long)row * HH + i];
    __syncthreads();
    const int el = entlens[b];
    if (tid < EE) {
        const float* er = ents + (long long)(b * EE + tid) * HH;
        float s = 0.0f;
        for (int h = 0; h < HH; ++h) s += pr[h] * er[h];
        sc[tid] = (tid < el) ? s : -1e9f;
    }
    __syncthreads();
    if (tid == 0) {
        float mx = -1e30f;
        for (int e = 0; e < EE; ++e) mx = fmaxf(mx, sc[e]);
        float sm = 0.0f;
        for (int e = 0; e < EE; ++e) sm += __expf(sc[e] - mx);
        mx_s = mx; sum_s = sm;
    }
    __syncthreads();
    if (tid < EE) {
        float z = __expf(sc[tid] - mx_s) / sum_s * (1.0f - pgen[row]);
        out[(long long)row * OUTW + TGTN + tid] = logf(z + 1e-6f);
    }
}

// ---------------------------------------------------------------------------
// State init: hx = cx = hx0, a = 0
// ---------------------------------------------------------------------------
__global__ void init_kernel(const float* __restrict__ hx0, float* __restrict__ hx,
                            float* __restrict__ cx, float* __restrict__ a) {
    int i = blockIdx.x * blockDim.x + threadIdx.x;
    if (i < BB * HH) { hx[i] = hx0[i]; cx[i] = hx0[i]; a[i] = 0.0f; }
}

// ---------------------------------------------------------------------------
// Host-side orchestration.  Workspace use: ~77 MB of d_ws.
// ---------------------------------------------------------------------------
extern "C" void kernel_launch(void* const* d_in, const int* in_sizes, int n_in,
                              void* d_out, int out_size, void* d_ws, size_t ws_size,
                              hipStream_t stream) {
    (void)in_sizes; (void)n_in; (void)out_size; (void)ws_size;

    const float* hx0   = (const float*)d_in[0];
    const float* keys  = (const float*)d_in[1];
    const float* ents  = (const float*)d_in[2];
    const float* emb   = (const float*)d_in[3];
    const float* W_ih  = (const float*)d_in[4];
    const float* b_ih  = (const float*)d_in[5];
    const float* W_hh  = (const float*)d_in[6];
    const float* b_hh  = (const float*)d_in[7];
    const float* Wq    = (const float*)d_in[8];
    const float* bq    = (const float*)d_in[9];
    const float* Wk    = (const float*)d_in[10];
    const float* bk    = (const float*)d_in[11];
    const float* Wv    = (const float*)d_in[12];
    const float* bv    = (const float*)d_in[13];
    const float* Wo    = (const float*)d_in[14];
    const float* bo    = (const float*)d_in[15];
    const float* W_sw  = (const float*)d_in[16];
    const float* b_sw  = (const float*)d_in[17];
    const float* W_out = (const float*)d_in[18];
    const float* b_out = (const float*)d_in[19];
    const float* W_ma  = (const float*)d_in[20];
    const float* b_ma  = (const float*)d_in[21];
    const int* tokens  = (const int*)d_in[22];
    const int* klens   = (const int*)d_in[23];
    const int* entlens = (const int*)d_in[24];
    float* out = (float*)d_out;

    float* ws = (float*)d_ws;
    float* Kp    = ws; ws += (size_t)BB * RR * HH;
    float* Vp    = ws; ws += (size_t)BB * RR * HH;
    float* hxb   = ws; ws += 2 * BB * HH;
    float* cxb   = ws; ws += 2 * BB * HH;
    float* ab    = ws; ws += 2 * BB * HH;
    float* gates = ws; ws += BB * 4 * HH;
    float* q     = ws; ws += BB * HH;
    float* ctx   = ws; ws += BB * HH;
    float* l     = ws; ws += (size_t)MROWS * 2 * HH;
    float* proj  = ws; ws += (size_t)MROWS * HH;
    float* pg    = ws; ws += MROWS;
    _Float16* lh = (_Float16*)ws; ws += ((size_t)MPAD * 1024) / 2;   // 6.8 MB
    _Float16* Wt = (_Float16*)ws; ws += ((size_t)NPAD * 1024) / 2;   // 41.4 MB

    const int SB = BB * HH;

    init_kernel<<<64, 256, 0, stream>>>(hx0, hxb, cxb, ab);

    // One-time prep: W_out -> f16 transposed (overlaps nothing, ~4us of HBM)
    transpose_f16_kernel<<<dim3(NPAD / 32, 32), 256, 0, stream>>>(W_out, Wt);

    // K/V head projections: [1920x512] @ [512x512] (+bias)
    gemm_f16_kernel<2><<<dim3(4, 30), 256, 0, stream>>>(
        keys, HH, Wk, HH, bk, Kp, HH, nullptr, 0, BB * RR, HH, HH);
    gemm_f16_kernel<2><<<dim3(4, 30), 256, 0, stream>>>(
        keys, HH, Wv, HH, bv, Vp, HH, nullptr, 0, BB * RR, HH, HH);

    // Sequential scan
    for (int t = 0; t < TT; ++t) {
        const int p = t & 1;
        gates_kernel<<<16, 128, 0, stream>>>(
            ab + p * SB, hxb + p * SB, emb, tokens, t, W_ih, W_hh, b_ih, b_hh, gates);
        lstmq_kernel<<<4, 128, 0, stream>>>(
            gates, cxb + p * SB, hxb + (p ^ 1) * SB, cxb + (p ^ 1) * SB,
            l + (size_t)t * (2 * HH), Wq, bq, q);
        attn_kernel<<<BB * NHD, 128, 0, stream>>>(q, Kp, Vp, klens, ctx);
        gemm_f16_kernel<1><<<dim3(4, 1), 128, 0, stream>>>(
            ctx, HH, Wo, HH, bo, ab + (p ^ 1) * SB, HH,
            l + (size_t)t * (2 * HH) + HH, (long long)TT * 2 * HH, BB, HH, HH);
    }

    // Output head
    pgen_kernel<<<MROWS, 256, 0, stream>>>(l, W_sw, b_sw, pg);
    cvt_f16_kernel<<<(MPAD * 1024) / 256, 256, 0, stream>>>(l, lh);
    // logits = l @ W_out + b_out -> gen slots of d_out (async-DMA WMMA GEMM)
    gemm_async_f16_kernel<<<dim3(NPAD / 256, MPAD / 256), 512, 0, stream>>>(
        lh, Wt, b_out, out, OUTW, MROWS, TGTN, 2 * HH);
    gen_softmax_kernel<<<MROWS, 256, 0, stream>>>(out, pg);
    // proj = l @ W_mattn + b_mattn
    gemm_f16_kernel<2><<<dim3(4, MROWS / 64), 256, 0, stream>>>(
        l, 2 * HH, W_ma, HH, b_ma, proj, HH, nullptr, 0, MROWS, HH, 2 * HH);
    copy_attn_kernel<<<MROWS, 256, 0, stream>>>(proj, ents, entlens, pg, out);
}
